// ResidualODE_11115375362893
// MI455X (gfx1250) — matrix operand order repaired and checked by reading, not compiled, not run on hardware
//
#include <hip/hip_runtime.h>
#include <hip/hip_bf16.h>

typedef __attribute__((ext_vector_type(16))) _Float16 v16h;
typedef __attribute__((ext_vector_type(8)))  float    v8f;

#define C_CH 64
#define HDIM 56
#define WDIM 56
#define XPAD 68                       // x halo: px up to 63 + kx 2 + 1 halo -> 66, pad to 68
#define ROWS_PER_BLK 2
#define STG_ROWS (ROWS_PER_BLK + 2)   // 4 staged rows
#define STAGE_ELEMS (STG_ROWS * XPAD * C_CH)   // 17,408 f16 = 34,816 B LDS
#define HW (HDIM * WDIM)              // 3136
#define NTOT (64ull * C_CH * HDIM * WDIM)     // 12,845,056 elements
#define WPACK_ELEMS (9 * 2 * 4 * 32 * 16)     // 36,864 f16 per weight set

// ---------------------------------------------------------------------------
// Pack w[Cout=64][Cin=64][3][3] fp32 -> f16 in the lane/element order this kernel uses for
// the 16-bit A-matrix fragment (16x32, contiguous k, same order as the B loads):
//   lane L: M = L%16 ; base = (L/16)*16
//   element i : K = base + i
// Layout: wp[((tap*2+chunk)*4 + co_tile)*512 + lane*16 + i]
// ---------------------------------------------------------------------------
__global__ __launch_bounds__(256) void pack_weights(const float* __restrict__ w,
                                                    _Float16* __restrict__ wp) {
    int idx = blockIdx.x * 256 + threadIdx.x;
    if (idx >= WPACK_ELEMS) return;
    int i       = idx & 15;
    int lane    = (idx >> 4) & 31;
    int co_tile = (idx >> 9) & 3;
    int tc      = idx >> 11;          // 0..17 = tap*2 + chunk
    int chunk   = tc & 1;
    int tap     = tc >> 1;
    int base    = (lane >> 4) * 16;
    int k       = base + i;
    int ci      = chunk * 32 + k;
    int co      = co_tile * 16 + (lane & 15);
    wp[idx] = (_Float16)w[(co * C_CH + ci) * 9 + tap];
}

// ---------------------------------------------------------------------------
// Per-(b,c) GroupNorm stats: mean and rstd over HxW (biased variance, eps=1e-5)
// ---------------------------------------------------------------------------
__global__ __launch_bounds__(256) void gn_stats(const float* __restrict__ src,
                                                float* __restrict__ mu,
                                                float* __restrict__ rstd) {
    int bc = blockIdx.x;                          // 0 .. B*C-1
    const float* p = src + (long)bc * HW;
    float s = 0.f, s2 = 0.f;
    for (int i = threadIdx.x; i < HW; i += 256) {
        float v = p[i];
        s += v; s2 += v * v;
    }
    __shared__ float sh1[256], sh2[256];
    sh1[threadIdx.x] = s; sh2[threadIdx.x] = s2;
    __syncthreads();
    for (int st = 128; st > 0; st >>= 1) {
        if (threadIdx.x < st) {
            sh1[threadIdx.x] += sh1[threadIdx.x + st];
            sh2[threadIdx.x] += sh2[threadIdx.x + st];
        }
        __syncthreads();
    }
    if (threadIdx.x == 0) {
        float m   = sh1[0] * (1.0f / HW);
        float var = sh2[0] * (1.0f / HW) - m * m;
        mu[bc]   = m;
        rstd[bc] = rsqrtf(var + 1e-5f);
    }
}

// ---------------------------------------------------------------------------
// Fused GroupNorm + ReLU + 3x3 conv (implicit GEMM on WMMA f16->f32).
// grid = (HDIM/2, B). Block = 256 threads = 8 waves.
// Stages 4 input rows; each wave owns (co_tile = wave&3, out-row = wave>>2)
// and computes FOUR 16x16 pixel tiles -> 4 independent accumulator chains,
// each A fragment reused 4x. 72 WMMA per wave per block.
// ---------------------------------------------------------------------------
__global__ __launch_bounds__(256, 2)
void gn_relu_conv3x3(const float* __restrict__ src,
                     const float* __restrict__ mu,   const float* __restrict__ rstd,
                     const float* __restrict__ gamma, const float* __restrict__ beta,
                     const _Float16* __restrict__ wpack,
                     const float* __restrict__ residual,   // may be nullptr
                     float* __restrict__ dst) {
    __shared__ alignas(32) _Float16 stage[STAGE_ELEMS];   // [row 0..3][x 0..67][ci 0..63]

    const int y0  = blockIdx.x * ROWS_PER_BLK;
    const int b   = blockIdx.y;
    const int tid = threadIdx.x;

    // ---- Phase 1: stage GN+ReLU(src) rows y0-1 .. y0+2 (with halo) as f16 ----
    for (int idx = tid; idx < STAGE_ELEMS; idx += 256) {
        int xl  = idx % XPAD;            // fastest: coalesced global reads
        int t2  = idx / XPAD;
        int ci  = t2 & 63;
        int row = t2 >> 6;               // 0..3
        int gy  = y0 + row - 1;
        int gx  = xl - 1;
        float v = 0.f;
        if ((unsigned)gy < HDIM && (unsigned)gx < WDIM) {
            int bc  = b * C_CH + ci;
            float x = src[((long)bc * HDIM + gy) * WDIM + gx];
            x = (x - mu[bc]) * rstd[bc] * gamma[ci] + beta[ci];
            v = x > 0.f ? x : 0.f;
        }
        stage[(row * XPAD + xl) * C_CH + ci] = (_Float16)v;
    }
    __syncthreads();

    // ---- Phase 2: WMMA implicit GEMM ----
    const int wave    = tid >> 5;
    const int lane    = tid & 31;
    const int co_tile = wave & 3;        // 4 tiles of 16 output channels
    const int rloc    = wave >> 2;       // output row within block: 0..1
    const int laneM   = lane & 15;
    const int laneH   = lane >> 4;

    v8f acc0 = {}, acc1 = {}, acc2 = {}, acc3 = {};
    const _Float16* wbase = wpack + co_tile * 512 + lane * 16;

#pragma unroll
    for (int tap = 0; tap < 9; ++tap) {
        const int ky = tap / 3, kx = tap % 3;
#pragma unroll
        for (int ch = 0; ch < 2; ++ch) {
            v16h afrag = *(const v16h*)(wbase + (tap * 2 + ch) * 2048);
            const int ci0 = ch * 32 + laneH * 16;
            const _Float16* bp =
                &stage[(((rloc + ky) * XPAD) + (laneM + kx)) * C_CH + ci0];
            v16h b0 = *(const v16h*)(bp);
            v16h b1 = *(const v16h*)(bp + 16 * C_CH);
            v16h b2 = *(const v16h*)(bp + 32 * C_CH);
            v16h b3 = *(const v16h*)(bp + 48 * C_CH);
            acc0 = __builtin_amdgcn_wmma_f32_16x16x32_f16(
                false, afrag, false, b0, (short)0, acc0, false, false);
            acc1 = __builtin_amdgcn_wmma_f32_16x16x32_f16(
                false, afrag, false, b1, (short)0, acc1, false, false);
            acc2 = __builtin_amdgcn_wmma_f32_16x16x32_f16(
                false, afrag, false, b2, (short)0, acc2, false, false);
            acc3 = __builtin_amdgcn_wmma_f32_16x16x32_f16(
                false, afrag, false, b3, (short)0, acc3, false, false);
        }
    }

    // ---- Store (C/D layout: lanes 0-15 -> M=r, lanes 16-31 -> M=8+r) ----
    const int coBase = co_tile * 16 + laneH * 8;
    const int yOut   = y0 + rloc;
#pragma unroll
    for (int r = 0; r < 8; ++r) {
        const int  co    = coBase + r;
        const long oBase = ((long)(b * C_CH + co) * HDIM + yOut) * WDIM;
        const int  px    = laneM;       // tile nt adds nt*16
        {
            long o = oBase + px;        // nt = 0 (px 0..15 always valid)
            float v = acc0[r];
            if (residual) v += residual[o];
            dst[o] = v;
        }
        {
            long o = oBase + px + 16;   // nt = 1
            float v = acc1[r];
            if (residual) v += residual[o];
            dst[o] = v;
        }
        {
            long o = oBase + px + 32;   // nt = 2
            float v = acc2[r];
            if (residual) v += residual[o];
            dst[o] = v;
        }
        if (px + 48 < WDIM) {           // nt = 3 (only px 48..55 valid)
            long o = oBase + px + 48;
            float v = acc3[r];
            if (residual) v += residual[o];
            dst[o] = v;
        }
    }
}

// ---------------------------------------------------------------------------
// Fused RK4 stage: single pass over k ->
//   zt  = z + a * k
//   acc = beta * acc + w * k          (beta = 0 initializes acc from k1)
// ---------------------------------------------------------------------------
__global__ __launch_bounds__(256) void rk4_stage(float4* __restrict__ zt,
                                                 float4* __restrict__ acc,
                                                 const float4* __restrict__ z,
                                                 const float4* __restrict__ k,
                                                 float a, float wgt, float betac,
                                                 int n4) {
    int i = blockIdx.x * 256 + threadIdx.x;
    if (i < n4) {
        float4 zv = z[i], kv = k[i], av = acc[i];
        float4 zo, ao;
        zo.x = zv.x + a * kv.x; zo.y = zv.y + a * kv.y;
        zo.z = zv.z + a * kv.z; zo.w = zv.w + a * kv.w;
        ao.x = betac * av.x + wgt * kv.x; ao.y = betac * av.y + wgt * kv.y;
        ao.z = betac * av.z + wgt * kv.z; ao.w = betac * av.w + wgt * kv.w;
        zt[i]  = zo;
        acc[i] = ao;
    }
}

// Final RK4 combine: z += s * (acc + k4)   (acc holds k1 + 2k2 + 2k3)
__global__ __launch_bounds__(256) void rk4_final(float4* __restrict__ z,
                                                 const float4* __restrict__ acc,
                                                 const float4* __restrict__ k,
                                                 float s, int n4) {
    int i = blockIdx.x * 256 + threadIdx.x;
    if (i < n4) {
        float4 zv = z[i], av = acc[i], kv = k[i];
        float4 r;
        r.x = zv.x + s * (av.x + kv.x); r.y = zv.y + s * (av.y + kv.y);
        r.z = zv.z + s * (av.z + kv.z); r.w = zv.w + s * (av.w + kv.w);
        z[i] = r;
    }
}

// ---------------------------------------------------------------------------
// Host side
// ---------------------------------------------------------------------------
static void odefunc_launch(const float* zin, float* kout, float* h1,
                           float* mu, float* rstd,
                           const _Float16* wp1, const _Float16* wp2,
                           const float* g1, const float* b1,
                           const float* g2, const float* b2,
                           hipStream_t stream) {
    dim3 cg(HDIM / ROWS_PER_BLK, 64);
    gn_stats<<<64 * C_CH, 256, 0, stream>>>(zin, mu, rstd);
    gn_relu_conv3x3<<<cg, 256, 0, stream>>>(zin, mu, rstd, g1, b1, wp1, nullptr, h1);
    gn_stats<<<64 * C_CH, 256, 0, stream>>>(h1, mu, rstd);
    gn_relu_conv3x3<<<cg, 256, 0, stream>>>(h1, mu, rstd, g2, b2, wp2, zin, kout);
}

extern "C" void kernel_launch(void* const* d_in, const int* in_sizes, int n_in,
                              void* d_out, int out_size, void* d_ws, size_t ws_size,
                              hipStream_t stream) {
    const float* x  = (const float*)d_in[0];
    const float* g1 = (const float*)d_in[1];
    const float* b1 = (const float*)d_in[2];
    const float* w1 = (const float*)d_in[3];
    const float* g2 = (const float*)d_in[4];
    const float* b2 = (const float*)d_in[5];
    const float* w2 = (const float*)d_in[6];

    const size_t tensorBytes = NTOT * sizeof(float);          // 51,380,224 B
    const size_t statBytes   = 64 * C_CH * sizeof(float);     // 16 KB
    const size_t wpBytes     = WPACK_ELEMS * sizeof(_Float16);// 72 KB

    char* ws = (char*)d_ws;
    size_t off = 0;
    float* z   = (float*)(ws + off); off += tensorBytes;
    float* zt  = (float*)(ws + off); off += tensorBytes;
    float* h1  = (float*)(ws + off); off += tensorBytes;
    float* kb  = (float*)(ws + off); off += tensorBytes;
    float* acc = (float*)(ws + off); off += tensorBytes;
    float* mu  = (float*)(ws + off); off += statBytes;
    float* rs  = (float*)(ws + off); off += statBytes;
    _Float16* wp1 = (_Float16*)(ws + off); off += wpBytes;
    _Float16* wp2 = (_Float16*)(ws + off); off += wpBytes;
    if (off > ws_size) return;   // workspace too small; nothing safe to do

    // One-time (per launch, deterministic) weight packing
    const int pwBlocks = (WPACK_ELEMS + 255) / 256;
    pack_weights<<<pwBlocks, 256, 0, stream>>>(w1, wp1);
    pack_weights<<<pwBlocks, 256, 0, stream>>>(w2, wp2);

    hipMemcpyAsync(z, x, tensorBytes, hipMemcpyDeviceToDevice, stream);

    const float dt = 1.0f / 8.0f;
    const int   n4 = (int)(NTOT / 4);
    const int   eb = (n4 + 255) / 256;

    for (int step = 0; step < 8; ++step) {
        // k1
        odefunc_launch(z, kb, h1, mu, rs, wp1, wp2, g1, b1, g2, b2, stream);
        rk4_stage<<<eb, 256, 0, stream>>>((float4*)zt, (float4*)acc,
                                          (const float4*)z, (const float4*)kb,
                                          0.5f * dt, 1.f, 0.f, n4);
        // k2
        odefunc_launch(zt, kb, h1, mu, rs, wp1, wp2, g1, b1, g2, b2, stream);
        rk4_stage<<<eb, 256, 0, stream>>>((float4*)zt, (float4*)acc,
                                          (const float4*)z, (const float4*)kb,
                                          0.5f * dt, 2.f, 1.f, n4);
        // k3
        odefunc_launch(zt, kb, h1, mu, rs, wp1, wp2, g1, b1, g2, b2, stream);
        rk4_stage<<<eb, 256, 0, stream>>>((float4*)zt, (float4*)acc,
                                          (const float4*)z, (const float4*)kb,
                                          dt, 2.f, 1.f, n4);
        // k4 ; z += (dt/6) * (k1 + 2k2 + 2k3 + k4)
        odefunc_launch(zt, kb, h1, mu, rs, wp1, wp2, g1, b1, g2, b2, stream);
        rk4_final<<<eb, 256, 0, stream>>>((float4*)z, (const float4*)acc,
                                          (const float4*)kb, dt / 6.f, n4);
    }

    hipMemcpyAsync(d_out, z, tensorBytes, hipMemcpyDeviceToDevice, stream);
}
